// Attention_867583394375
// MI455X (gfx1250) — matrix operand order repaired and checked by reading, not compiled, not run on hardware
//
#include <hip/hip_runtime.h>

// ---------------------------------------------------------------------------
// MI455X (gfx1250) attention pipeline, bf16 WMMA everywhere, f32 softmax/BN.
// Requires ws_size >= ~101 MB.
// ---------------------------------------------------------------------------

typedef unsigned short u16;
typedef __attribute__((ext_vector_type(4)))  unsigned int u32x4;
typedef __attribute__((ext_vector_type(4)))  float        f32x4;
typedef __attribute__((ext_vector_type(8)))  float        v8f;
typedef __attribute__((ext_vector_type(16))) __bf16       v16bf;

static __device__ __forceinline__ u16 f2bf(float f) {
    unsigned int u = __float_as_uint(f);
    u += 0x7fffu + ((u >> 16) & 1u);   // round-to-nearest-even
    return (u16)(u >> 16);
}

union FragAB { v16bf v; u32x4 q[2]; };
union Pack16 { u16 h[16]; u32x4 q[2]; };
union Pack8  { u16 h[8];  u32x4 q; };

// ---------------------------------------------------------------------------
// Generic tiled bf16 WMMA GEMM:  C[M,N] = A[M,K] @ B[K,N]   (per-batch z)
//   BSRC = 0 : B source is bf16 row-major [K,N]
//   BSRC = 1 : B source is f32  row-major [N,K]  (i.e. B given transposed;
//              used for out = attn @ V with attn read as B^T, converted on the fly)
//   OUT  = 0 : bf16 row-major [M,N]
//   OUT  = 1 : bf16 transposed store -> [N,M]   (q projection)
//   OUT  = 2 : f32 row-major, scaled (attention logits, non-temporal store)
//   OUT  = 3 : f32 row-major + per-row (channel) bias
// Tile: BM=128, BN=128, BK=32.  256 threads = 8 waves, each wave 2x4 WMMA tiles.
// ---------------------------------------------------------------------------
template <int BSRC, int OUT>
__global__ __launch_bounds__(256) void gemm_bf16_wmma(
    const u16* __restrict__ A, long long strideA, int lda,
    const void* __restrict__ Bsrc, long long strideB, int ldb,
    void* __restrict__ Cdst, long long strideC, int ldc,
    int K, float scale, const float* __restrict__ bias)
{
    constexpr int LDT = 40;                       // padded LDS row (elems), 80B = 16B-aligned
    __shared__ __align__(16) u16 sA[128 * LDT];   // [m][k]
    __shared__ __align__(16) u16 sB[128 * LDT];   // [n][k]  (B staged transposed)

    const int tid  = threadIdx.x;
    const int lane = tid & 31;
    const int wave = tid >> 5;
    const int wm   = wave >> 1;                   // 0..3 : 32 rows each
    const int wn   = wave & 1;                    // 0..1 : 64 cols each
    const int bm   = blockIdx.y * 128;
    const int bn   = blockIdx.x * 128;

    const u16*   Ab  = A + (long long)blockIdx.z * strideA;
    const u16*   B16 = (const u16*)Bsrc + (long long)blockIdx.z * strideB;
    const float* B32 = (const float*)Bsrc + (long long)blockIdx.z * strideB;

    v8f acc[2][4] = {};

    for (int k0 = 0; k0 < K; k0 += 32) {
        // ---- stage A tile [128 x 32] (row-major source, coalesced b128 loads)
        {
            const int mm = tid >> 1;
            const int kk = (tid & 1) * 16;
            const u16* src = Ab + (long long)(bm + mm) * lda + k0 + kk;
            u32x4 p0 = *(const u32x4*)src;
            u32x4 p1 = *(const u32x4*)(src + 8);
            *(u32x4*)&sA[mm * LDT + kk]     = p0;
            *(u32x4*)&sA[mm * LDT + kk + 8] = p1;
        }
        // ---- stage B tile -> sB[n][k]
        if (BSRC == 0) {
            const int kk = tid >> 3;              // 0..31
            const int ns = (tid & 7) * 16;        // 0..112
            const u16* src = B16 + (long long)(k0 + kk) * ldb + bn + ns;
            Pack16 pk;
            pk.q[0] = *(const u32x4*)src;
            pk.q[1] = *(const u32x4*)(src + 8);
#pragma unroll
            for (int i = 0; i < 16; ++i) sB[(ns + i) * LDT + kk] = pk.h[i];
        } else {
            const int nn = tid >> 1;              // 0..127
            const int kk = (tid & 1) * 16;
            const float* src = B32 + (long long)(bn + nn) * ldb + k0 + kk;
            Pack16 pk;
#pragma unroll
            for (int q4 = 0; q4 < 4; ++q4) {
                f32x4 f = __builtin_nontemporal_load((const f32x4*)src + q4);
                pk.h[q4 * 4 + 0] = f2bf(f[0]);
                pk.h[q4 * 4 + 1] = f2bf(f[1]);
                pk.h[q4 * 4 + 2] = f2bf(f[2]);
                pk.h[q4 * 4 + 3] = f2bf(f[3]);
            }
            *(u32x4*)&sB[nn * LDT + kk]     = pk.q[0];
            *(u32x4*)&sB[nn * LDT + kk + 8] = pk.q[1];
        }
        __syncthreads();

        // ---- build fragments (ISA 7.12.2 layouts) and issue 8 WMMAs/wave
        FragAB af[2];
        const int ar = lane & 15;
        const int ak = (lane >> 4) * 8;           // K 0..7 / 8..15 halves
#pragma unroll
        for (int i = 0; i < 2; ++i) {
            const u16* base = &sA[(wm * 32 + i * 16 + ar) * LDT + ak];
            af[i].q[0] = *(const u32x4*)base;         // K = ak .. ak+7
            af[i].q[1] = *(const u32x4*)(base + 16);  // K = ak+16 .. ak+23
        }
        FragAB bf[4];
        const int bk = (lane >> 4) * 8;           // K 0..7 / 8..15 halves
#pragma unroll
        for (int j = 0; j < 4; ++j) {
            const u16* base = &sB[(wn * 64 + j * 16 + (lane & 15)) * LDT + bk];
            bf[j].q[0] = *(const u32x4*)base;
            bf[j].q[1] = *(const u32x4*)(base + 16);
        }
#pragma unroll
        for (int i = 0; i < 2; ++i)
#pragma unroll
            for (int j = 0; j < 4; ++j)
                acc[i][j] = __builtin_amdgcn_wmma_f32_16x16x32_bf16(
                    false, af[i].v, false, bf[j].v, (short)0, acc[i][j],
                    false, false);
        __syncthreads();
    }

    // ---- store D (C/D layout: VGPR r -> m = r + 8*(lane>>4), n = lane&15)
    const long long cz = (long long)blockIdx.z * strideC;
#pragma unroll
    for (int i = 0; i < 2; ++i) {
#pragma unroll
        for (int j = 0; j < 4; ++j) {
            const int mbase = bm + wm * 32 + i * 16 + ((lane >> 4) << 3);
            const int n     = bn + wn * 64 + j * 16 + (lane & 15);
            if (OUT == 0) {
                u16* Cp = (u16*)Cdst + cz;
#pragma unroll
                for (int r = 0; r < 8; ++r)
                    Cp[(long long)(mbase + r) * ldc + n] = f2bf(acc[i][j][r]);
            } else if (OUT == 1) {
                u16* Cp = (u16*)Cdst + cz;
                Pack8 pk;
#pragma unroll
                for (int r = 0; r < 8; ++r) pk.h[r] = f2bf(acc[i][j][r]);
                *(u32x4*)&Cp[(long long)n * ldc + mbase] = pk.q;  // 8 contiguous m
            } else if (OUT == 2) {
                float* Cp = (float*)Cdst + cz;
#pragma unroll
                for (int r = 0; r < 8; ++r)
                    __builtin_nontemporal_store(acc[i][j][r] * scale,
                        &Cp[(long long)(mbase + r) * ldc + n]);
            } else {
                float* Cp = (float*)Cdst + cz;
#pragma unroll
                for (int r = 0; r < 8; ++r)
                    Cp[(long long)(mbase + r) * ldc + n] =
                        acc[i][j][r] + bias[mbase + r];
            }
        }
    }
}

// ---------------------------------------------------------------------------
// In-place row softmax over 4096-wide f32 rows (one workgroup per row).
// ---------------------------------------------------------------------------
__global__ __launch_bounds__(256) void softmax_rows(float* __restrict__ attn)
{
    const long long row = blockIdx.x;
    f32x4* pv = (f32x4*)(attn + row * 4096);
    const int tid = threadIdx.x, lane = tid & 31, wv = tid >> 5;

    f32x4 d[4];
    float mx = -3.4e38f;
#pragma unroll
    for (int k = 0; k < 4; ++k) {
        d[k] = __builtin_nontemporal_load(pv + tid + 256 * k);
#pragma unroll
        for (int e = 0; e < 4; ++e) mx = fmaxf(mx, d[k][e]);
    }
#pragma unroll
    for (int off = 16; off > 0; off >>= 1) mx = fmaxf(mx, __shfl_xor(mx, off, 32));
    __shared__ float red[8];
    if (lane == 0) red[wv] = mx;
    __syncthreads();
    float rowmax = red[0];
#pragma unroll
    for (int w = 1; w < 8; ++w) rowmax = fmaxf(rowmax, red[w]);
    __syncthreads();

    float s = 0.f;
#pragma unroll
    for (int k = 0; k < 4; ++k)
#pragma unroll
        for (int e = 0; e < 4; ++e) { float v = __expf(d[k][e] - rowmax); d[k][e] = v; s += v; }
#pragma unroll
    for (int off = 16; off > 0; off >>= 1) s += __shfl_xor(s, off, 32);
    if (lane == 0) red[wv] = s;
    __syncthreads();
    float tot = 0.f;
#pragma unroll
    for (int w = 0; w < 8; ++w) tot += red[w];
    const float inv = 1.0f / tot;
#pragma unroll
    for (int k = 0; k < 4; ++k) {
        f32x4 o = d[k] * inv;
        __builtin_nontemporal_store(o, pv + tid + 256 * k);
    }
}

// ---------------------------------------------------------------------------
// BatchNorm training-mode stats: per channel mean + rsqrt(var+eps).
// y layout: [B=8][C=256][N=4096] f32. One workgroup per channel.
// ---------------------------------------------------------------------------
__global__ __launch_bounds__(256) void bn_stats_kernel(
    const float* __restrict__ y, float* __restrict__ stats)
{
    const int o = blockIdx.x, tid = threadIdx.x, lane = tid & 31, wv = tid >> 5;
    float s = 0.f, s2 = 0.f;
    for (int b = 0; b < 8; ++b) {
        const float* p = y + ((long long)b * 256 + o) * 4096;
        for (int n = tid; n < 4096; n += 256) { float v = p[n]; s += v; s2 += v * v; }
    }
#pragma unroll
    for (int off = 16; off > 0; off >>= 1) { s += __shfl_xor(s, off, 32); s2 += __shfl_xor(s2, off, 32); }
    __shared__ float rs[8], rq[8];
    if (lane == 0) { rs[wv] = s; rq[wv] = s2; }
    __syncthreads();
    if (tid == 0) {
        float S = 0.f, S2 = 0.f;
#pragma unroll
        for (int w = 0; w < 8; ++w) { S += rs[w]; S2 += rq[w]; }
        const float mean = S * (1.0f / 32768.0f);
        const float var  = S2 * (1.0f / 32768.0f) - mean * mean;
        stats[o]       = mean;
        stats[256 + o] = rsqrtf(var + 1e-5f);
    }
}

__global__ __launch_bounds__(256) void bn_apply_kernel(
    const float* __restrict__ y, const float* __restrict__ stats,
    const float* __restrict__ gamma, const float* __restrict__ beta,
    float* __restrict__ act)
{
    const long long base = ((long long)blockIdx.x * 256 + threadIdx.x) * 4;
    const int o = (int)((base >> 12) & 255);
    const float a = gamma[o] * stats[256 + o];
    const float c = beta[o] - stats[o] * a;
    f32x4 v = *(const f32x4*)(y + base);
    f32x4 r;
#pragma unroll
    for (int e = 0; e < 4; ++e) {
        float t = v[e] * a + c;
        r[e] = (t >= 0.f) ? t : 0.01f * t;
    }
    *(f32x4*)(act + base) = r;
}

__global__ __launch_bounds__(256) void cvt_f32_bf16(
    const float* __restrict__ src, u16* __restrict__ dst, long long n)
{
    const long long i = ((long long)blockIdx.x * 256 + threadIdx.x) * 4;
    if (i >= n) return;
    f32x4 f = *(const f32x4*)(src + i);
    Pack8 pk;
    pk.h[0] = f2bf(f[0]); pk.h[1] = f2bf(f[1]);
    pk.h[2] = f2bf(f[2]); pk.h[3] = f2bf(f[3]);
    dst[i + 0] = pk.h[0]; dst[i + 1] = pk.h[1];
    dst[i + 2] = pk.h[2]; dst[i + 3] = pk.h[3];
}

// ---------------------------------------------------------------------------
extern "C" void kernel_launch(void* const* d_in, const int* in_sizes, int n_in,
                              void* d_out, int out_size, void* d_ws, size_t ws_size,
                              hipStream_t stream)
{
    (void)in_sizes; (void)n_in; (void)out_size; (void)ws_size;
    const float* x     = (const float*)d_in[0];   // [8,256,64,64]
    const float* Wqk   = (const float*)d_in[1];   // [512,256]
    const float* Wout  = (const float*)d_in[2];   // [256,256]
    const float* bout  = (const float*)d_in[3];   // [256]
    const float* gamma = (const float*)d_in[4];   // [256]
    const float* beta  = (const float*)d_in[5];   // [256]

    const long long B = 8, C = 256, N = 4096;
    float* act  = (float*)d_out;                  // [8,256,64,64]
    float* attn = act + B * C * N;                // [8,4096,4096]

    // Workspace layout (all offsets 256B-aligned; total ~101 MB).
    char* ws = (char*)d_ws;
    u16*   xb    = (u16*)(ws);                                   // [8,256,4096] bf16
    u16*   wqkb  = (u16*)(ws + 16777216);                        // [512,256]
    u16*   woutb = (u16*)(ws + 16777216 + 262144);               // [256,256]
    u16*   qrow  = (u16*)(ws + 16777216 + 262144 + 131072);      // [8,4096,256]
    u16*   kT    = qrow  + B * N * C;                            // [8,256,4096]
    u16*   outT  = kT    + B * C * N;                            // [8,256,4096]
    float* yws   = (float*)((char*)(outT + B * C * N));          // [8,256,4096] f32
    float* stats = yws + B * C * N;                              // [512]

    // 0) f32 -> bf16 conversions
    cvt_f32_bf16<<<8192, 256, 0, stream>>>(x,    xb,    B * C * N);
    cvt_f32_bf16<<<128,  256, 0, stream>>>(Wqk,  wqkb,  512 * 256);
    cvt_f32_bf16<<<64,   256, 0, stream>>>(Wout, woutb, 256 * 256);

    // 1) q = (Wqk_q @ x[b])^T  -> qrow [b][n][d]   (transposed store)
    gemm_bf16_wmma<0, 1><<<dim3(32, 2, 8), 256, 0, stream>>>(
        wqkb, 0LL, 256, xb, C * N, 4096, qrow, N * C, 256, 256, 1.f, nullptr);
    //    k^T = Wqk_k @ x[b]    -> kT [b][d][n]
    gemm_bf16_wmma<0, 0><<<dim3(32, 2, 8), 256, 0, stream>>>(
        wqkb + 256 * 256, 0LL, 256, xb, C * N, 4096, kT, C * N, 4096, 256, 1.f, nullptr);

    // 2) dots = q @ k^T * inner^-0.5  -> attn slot (f32 logits, streaming)
    gemm_bf16_wmma<0, 2><<<dim3(32, 32, 8), 256, 0, stream>>>(
        qrow, N * C, 256, kT, C * N, 4096, attn, N * N, 4096, 256, 0.0625f, nullptr);

    // 3) softmax rows in place
    softmax_rows<<<(unsigned)(B * N), 256, 0, stream>>>(attn);

    // 4) out^T = x_bf @ attn^T  -> outT [b][c][n]   (attn read as B^T, f32->bf16)
    gemm_bf16_wmma<1, 0><<<dim3(32, 2, 8), 256, 0, stream>>>(
        xb, C * N, 4096, attn, N * N, 4096, outT, C * N, 4096, 4096, 1.f, nullptr);

    // 5) y = Wout @ out^T + bout  -> yws [b][o][n] f32
    gemm_bf16_wmma<0, 3><<<dim3(32, 2, 8), 256, 0, stream>>>(
        woutb, 0LL, 256, outT, C * N, 4096, yws, C * N, 4096, 256, 1.f, bout);

    // 6) BatchNorm (training stats) + LeakyReLU -> act
    bn_stats_kernel<<<256, 256, 0, stream>>>(yws, stats);
    bn_apply_kernel<<<8192, 256, 0, stream>>>(yws, stats, gamma, beta, act);
}
